// BRNN_19928648253951
// MI455X (gfx1250) — compile-verified, matrix operand order checked by reading, not run on hardware
//
#include <hip/hip_runtime.h>
#include <hip/hip_bf16.h>
#include <math.h>

// ---------------------------------------------------------------------------
// BRNN recurrence on MI455X (gfx1250), wave32 + v_wmma_f32_16x16x32_bf16.
// Weights (84MB fp32 -> 42MB bf16) fit in the 192MB L2 and are re-read each
// of 256 time steps; bf16 WMMA halves L2 traffic and uses the matrix pipes.
// Persistent kernel: 32 blocks x 128 threads (128 waves), software grid
// barrier between the serialized phases of the recurrence.
// ---------------------------------------------------------------------------

typedef __bf16 bf16_t;
typedef __attribute__((ext_vector_type(16))) __bf16 v16bf;
typedef __attribute__((ext_vector_type(8)))  float  v8f;

#define BATCH   32
#define TSTEPS  256
#define NIN     256
#define NHID    2048
#define NOUT    256
#define NPART   4
#define SPD     512            // NHID / NPART

#define NBLK    32             // persistent blocks (all co-resident)
#define TPB     128            // 4 waves/block -> 128 waves total
#define TILE_ELEMS 512         // one packed B tile: 32 lanes * 16 bf16

// ---- WMMA helpers ----------------------------------------------------------

__device__ __forceinline__ v8f wmma_bf16(v16bf a, v16bf b, v8f c) {
    // 8 args: (neg_a, A, neg_b, B, c_mod, C, reuse_a, reuse_b)
    return __builtin_amdgcn_wmma_f32_16x16x32_bf16(
        false, a, false, b, (short)0, c, false, false);
}

// A operand (16x32 bf16 tile) loaded straight from a row-major bf16 matrix.
// ISA layout: lane holds row (lane&15); lanes<16 carry K {0..7,16..23},
// lanes>=16 carry K {8..15,24..31} -> two contiguous 16B runs per lane.
__device__ __forceinline__ v16bf load_a(const bf16_t* base, int stride,
                                        int row0, int k0, int lane) {
    int row = row0 + (lane & 15);
    int kb  = k0 + ((lane >> 4) << 3);
    const bf16_t* p = base + (size_t)row * stride + kb;
    union { uint4 q[2]; v16bf v; } u;
    u.q[0] = *reinterpret_cast<const uint4*>(p);
    u.q[1] = *reinterpret_cast<const uint4*>(p + 16);
    return u.v;
}

// B operand from the pre-packed tile array: lane reads 32 contiguous bytes.
__device__ __forceinline__ v16bf load_bp(const bf16_t* pack, size_t tile,
                                         int lane) {
    const bf16_t* p = pack + tile * TILE_ELEMS + lane * 16;
    union { uint4 q[2]; v16bf v; } u;
    u.q[0] = reinterpret_cast<const uint4*>(p)[0];
    u.q[1] = reinterpret_cast<const uint4*>(p)[1];
    return u.v;
}

__device__ __forceinline__ float softplus_f(float v) {
    return v > 20.0f ? v : log1pf(expf(v));
}
__device__ __forceinline__ float softsign_f(float v) {
    return v / (1.0f + fabsf(v));
}

// Grid-wide barrier: monotonic counter, agent-scope release/acquire.
__device__ __forceinline__ void gbar(unsigned* ctr, unsigned& phase) {
    phase += NBLK;
    __syncthreads();
    if (threadIdx.x == 0) {
        __threadfence();
        __hip_atomic_fetch_add(ctr, 1u, __ATOMIC_RELEASE,
                               __HIP_MEMORY_SCOPE_AGENT);
        while (__hip_atomic_load(ctr, __ATOMIC_ACQUIRE,
                                 __HIP_MEMORY_SCOPE_AGENT) < phase) {
            __builtin_amdgcn_s_sleep(2);
        }
        __threadfence();
    }
    __syncthreads();
}

// ---- Prep: zero barrier, x [B,T,I] -> bf16 [T,B,I], init hidden ------------

__global__ void brnn_prep(const float* __restrict__ x,
                          const float* __restrict__ hi_W,
                          const float* __restrict__ hi_b,
                          bf16_t* __restrict__ xP,
                          float*  __restrict__ hidF,
                          bf16_t* __restrict__ hidB,
                          unsigned* __restrict__ ctr) {
    int idx = blockIdx.x * blockDim.x + threadIdx.x;
    int nthr = gridDim.x * blockDim.x;
    if (idx == 0) *ctr = 0u;
    const int totalX = BATCH * TSTEPS * NIN;
    for (int i = idx; i < totalX; i += nthr) {
        int b = i / (TSTEPS * NIN);
        int r = i % (TSTEPS * NIN);
        int t = r / NIN, ii = r % NIN;
        xP[((size_t)t * BATCH + b) * NIN + ii] = (bf16_t)x[i];
    }
    const int totalH = BATCH * NHID;
    for (int i = idx; i < totalH; i += nthr) {
        int j = i % NHID;
        float h = hi_W[j] + hi_b[j];
        hidF[i] = h;
        hidB[i] = (bf16_t)h;
    }
}

// ---- Pack W[K][N] (f32 row-major) into bf16 WMMA B tiles -------------------
// Packed order: tile = kt*(N/16)+nt ; within tile: lane-major, 16 halfs/lane.
__global__ void brnn_pack_b(const float* __restrict__ src,
                            bf16_t* __restrict__ dst, int K, int N) {
    int NT = N >> 4;
    int total = K * N;
    int nthr = gridDim.x * blockDim.x;
    for (int idx = blockIdx.x * blockDim.x + threadIdx.x; idx < total;
         idx += nthr) {
        int s    = idx & 15;
        int lane = (idx >> 4) & 31;
        int tile = idx >> 9;
        int nt = tile % NT, kt = tile / NT;
        int half = lane >> 4, nsub = lane & 15;
        int koff = half * 8 + (s < 8 ? s : s + 8);   // runs {0..7},{16..23}
        int k = kt * 32 + koff;
        int col = nt * 16 + nsub;
        dst[idx] = (bf16_t)src[(size_t)k * N + col];
    }
}

// ---- Persistent recurrence kernel ------------------------------------------

__global__ __launch_bounds__(TPB) void brnn_main(
    const float* __restrict__ enc_b, const float* __restrict__ rec_b,
    const float* __restrict__ dec_b,
    const bf16_t* __restrict__ xP,   const bf16_t* __restrict__ modP,
    const bf16_t* __restrict__ recP, const bf16_t* __restrict__ encP,
    const bf16_t* __restrict__ decP,
    float* __restrict__ hidF, bf16_t* __restrict__ hidB,
    bf16_t* __restrict__ gB,
    unsigned* __restrict__ ctr, float* __restrict__ out) {

    const int lane = threadIdx.x & 31;
    const int wave = (blockIdx.x * TPB + threadIdx.x) >> 5;   // 0..127
    const int nsub = lane & 15;
    const int half = lane >> 4;
    unsigned phase = 0;

    for (int t = 0; t < TSTEPS; ++t) {
        const bf16_t* xt = xP + (size_t)t * BATCH * NIN;

        for (int n = 0; n < NPART; ++n) {
            // ---- phase 1: tmp = hid @ mod_W[n];  g = softplus(tmp * hid)
            {
                const bf16_t* mp = modP + (size_t)n * NHID * NHID;
                const int nt = wave;                 // 128 N-tiles of 2048 cols
                const int jj = nt * 16 + nsub;
                v8f acc0 = {}, acc1 = {};
                #pragma unroll 4
                for (int kt = 0; kt < NHID / 32; ++kt) {
                    v16bf a0 = load_a(hidB, NHID, 0,  kt * 32, lane);
                    v16bf a1 = load_a(hidB, NHID, 16, kt * 32, lane);
                    v16bf b  = load_bp(mp, (size_t)kt * (NHID / 16) + nt, lane);
                    acc0 = wmma_bf16(a0, b, acc0);
                    acc1 = wmma_bf16(a1, b, acc1);
                }
                #pragma unroll
                for (int mt = 0; mt < 2; ++mt) {
                    v8f acc = mt ? acc1 : acc0;
                    #pragma unroll
                    for (int r = 0; r < 8; ++r) {
                        int m = mt * 16 + half * 8 + r;
                        float h = hidF[(size_t)m * NHID + jj];
                        float g = softplus_f(acc[r] * h);
                        gB[(size_t)m * NHID + jj] = (bf16_t)g;
                    }
                }
            }
            gbar(ctr, phase);

            // ---- phase 2: part = softsign(x_t@enc + g@rec + biases)
            if (wave < SPD / 16) {                   // 32 N-tiles of 512 cols
                const int nt = wave;
                const int jloc  = nt * 16 + nsub;
                const int jglob = n * SPD + jloc;
                float bias = enc_b[n * SPD + jloc] + rec_b[n * SPD + jloc];
                v8f acc0, acc1;
                #pragma unroll
                for (int r = 0; r < 8; ++r) { acc0[r] = bias; acc1[r] = bias; }

                const bf16_t* ep = encP + (size_t)n * NIN * SPD;
                #pragma unroll
                for (int kt = 0; kt < NIN / 32; ++kt) {
                    v16bf a0 = load_a(xt, NIN, 0,  kt * 32, lane);
                    v16bf a1 = load_a(xt, NIN, 16, kt * 32, lane);
                    v16bf b  = load_bp(ep, (size_t)kt * (SPD / 16) + nt, lane);
                    acc0 = wmma_bf16(a0, b, acc0);
                    acc1 = wmma_bf16(a1, b, acc1);
                }
                const bf16_t* rp = recP + (size_t)n * NHID * SPD;
                #pragma unroll 4
                for (int kt = 0; kt < NHID / 32; ++kt) {
                    v16bf a0 = load_a(gB, NHID, 0,  kt * 32, lane);
                    v16bf a1 = load_a(gB, NHID, 16, kt * 32, lane);
                    v16bf b  = load_bp(rp, (size_t)kt * (SPD / 16) + nt, lane);
                    acc0 = wmma_bf16(a0, b, acc0);
                    acc1 = wmma_bf16(a1, b, acc1);
                }
                #pragma unroll
                for (int mt = 0; mt < 2; ++mt) {
                    v8f acc = mt ? acc1 : acc0;
                    #pragma unroll
                    for (int r = 0; r < 8; ++r) {
                        int m = mt * 16 + half * 8 + r;
                        float p = softsign_f(acc[r]);
                        hidF[(size_t)m * NHID + jglob] = p;
                        hidB[(size_t)m * NHID + jglob] = (bf16_t)p;
                    }
                }
            }
            gbar(ctr, phase);
        }

        // ---- decode: out[t] = hid @ dec_W + dec_b  (waves 0..15)
        if (wave < NOUT / 16) {
            const int nt = wave;
            const int j = nt * 16 + nsub;
            float bias = dec_b[j];
            v8f acc0, acc1;
            #pragma unroll
            for (int r = 0; r < 8; ++r) { acc0[r] = bias; acc1[r] = bias; }
            #pragma unroll 4
            for (int kt = 0; kt < NHID / 32; ++kt) {
                v16bf a0 = load_a(hidB, NHID, 0,  kt * 32, lane);
                v16bf a1 = load_a(hidB, NHID, 16, kt * 32, lane);
                v16bf b  = load_bp(decP, (size_t)kt * (NOUT / 16) + nt, lane);
                acc0 = wmma_bf16(a0, b, acc0);
                acc1 = wmma_bf16(a1, b, acc1);
            }
            float* ot = out + (size_t)t * BATCH * NOUT;
            #pragma unroll
            for (int mt = 0; mt < 2; ++mt) {
                v8f acc = mt ? acc1 : acc0;
                #pragma unroll
                for (int r = 0; r < 8; ++r) {
                    int m = mt * 16 + half * 8 + r;
                    ot[(size_t)m * NOUT + j] = acc[r];
                }
            }
        }
        // Next step's hidden writes are fenced by the phase-2 barrier, which
        // decode waves reach only after finishing here -> no extra barrier.
    }

    // append final hidden [B, NHID] after outputs
    float* out2 = out + (size_t)TSTEPS * BATCH * NOUT;
    int gid = blockIdx.x * TPB + threadIdx.x;
    for (int i = gid; i < BATCH * NHID; i += NBLK * TPB) out2[i] = hidF[i];
}

// ---- Host launcher ---------------------------------------------------------

extern "C" void kernel_launch(void* const* d_in, const int* in_sizes, int n_in,
                              void* d_out, int out_size, void* d_ws,
                              size_t ws_size, hipStream_t stream) {
    const float* x     = (const float*)d_in[0];   // [B,T,I]
    const float* enc_W = (const float*)d_in[1];   // [4,256,512]
    const float* enc_b = (const float*)d_in[2];   // [4,512]
    const float* rec_W = (const float*)d_in[3];   // [4,2048,512]
    const float* rec_b = (const float*)d_in[4];   // [4,512]
    const float* mod_W = (const float*)d_in[5];   // [4,2048,2048]
    const float* dec_W = (const float*)d_in[6];   // [2048,256]
    const float* dec_b = (const float*)d_in[7];   // [256]
    const float* hi_W  = (const float*)d_in[8];   // [2048]
    const float* hi_b  = (const float*)d_in[9];   // [2048]
    float* out = (float*)d_out;

    // workspace carve-out (~49 MB)
    size_t off = 0;
    char* ws = (char*)d_ws;
    auto carve = [&](size_t bytes) -> void* {
        void* p = ws + off;
        off += (bytes + 255) & ~(size_t)255;
        return p;
    };
    unsigned* ctr  = (unsigned*)carve(256);
    float*  hidF = (float*) carve((size_t)BATCH * NHID * 4);
    bf16_t* hidB = (bf16_t*)carve((size_t)BATCH * NHID * 2);
    bf16_t* gB   = (bf16_t*)carve((size_t)BATCH * NHID * 2);
    bf16_t* xP   = (bf16_t*)carve((size_t)TSTEPS * BATCH * NIN * 2);
    bf16_t* modP = (bf16_t*)carve((size_t)NPART * NHID * NHID * 2);
    bf16_t* recP = (bf16_t*)carve((size_t)NPART * NHID * SPD * 2);
    bf16_t* encP = (bf16_t*)carve((size_t)NPART * NIN * SPD * 2);
    bf16_t* decP = (bf16_t*)carve((size_t)NHID * NOUT * 2);
    (void)ws_size; (void)in_sizes; (void)n_in; (void)out_size;

    brnn_prep<<<1024, 256, 0, stream>>>(x, hi_W, hi_b, xP, hidF, hidB, ctr);

    for (int p = 0; p < NPART; ++p) {
        brnn_pack_b<<<2048, 256, 0, stream>>>(
            mod_W + (size_t)p * NHID * NHID, modP + (size_t)p * NHID * NHID,
            NHID, NHID);
        brnn_pack_b<<<512, 256, 0, stream>>>(
            rec_W + (size_t)p * NHID * SPD, recP + (size_t)p * NHID * SPD,
            NHID, SPD);
        brnn_pack_b<<<128, 256, 0, stream>>>(
            enc_W + (size_t)p * NIN * SPD, encP + (size_t)p * NIN * SPD,
            NIN, SPD);
    }
    brnn_pack_b<<<512, 256, 0, stream>>>(dec_W, decP, NHID, NOUT);

    brnn_main<<<NBLK, TPB, 0, stream>>>(enc_b, rec_b, dec_b, xP, modP, recP,
                                        encP, decP, hidF, hidB, gB, ctr, out);
}
